// ExportableAttention_85152021610665
// MI455X (gfx1250) — matrix operand order repaired and checked by reading, not compiled, not run on hardware
//
#include <hip/hip_runtime.h>
#include <hip/hip_bf16.h>
#include <math.h>

typedef __bf16 bf16;
typedef __attribute__((ext_vector_type(16))) __bf16 v16bf;
typedef __attribute__((ext_vector_type(8)))  __bf16 bf16x8;
typedef __attribute__((ext_vector_type(8)))  float  v8f;

#define HIDDEN    1024
#define NUM_HEADS 16
#define HEAD_SIZE 64
#define SEQ       4096
#define BATCH     2
#define MTOT      (BATCH*SEQ)   // 8192

// ---------------- async global->LDS copy (CDNA5 path, with safe fallback) ---------
#if defined(__has_builtin)
#  if __has_builtin(__builtin_amdgcn_global_load_async_to_lds_b128)
#    define USE_ASYNC 1
#  endif
#endif
#ifndef USE_ASYNC
#  define USE_ASYNC 0
#endif

#if USE_ASYNC
typedef int v4i __attribute__((vector_size(16)));
typedef __attribute__((address_space(1))) v4i* gbl_v4i;
typedef __attribute__((address_space(3))) v4i* lds_v4i;
__device__ __forceinline__ void copy_b128(const bf16* g, bf16* l) {
    // emits global_load_async_to_lds_b128 (ASYNCcnt-tracked, no VGPR round trip)
    __builtin_amdgcn_global_load_async_to_lds_b128((gbl_v4i)g, (lds_v4i)l, 0, 0);
}
#  if __has_builtin(__builtin_amdgcn_s_wait_asynccnt)
#    define WAIT_ASYNC(n) __builtin_amdgcn_s_wait_asynccnt(n)
#  else
#    define WAIT_ASYNC(n) asm volatile("s_wait_asynccnt %0" :: "i"(n) : "memory")
#  endif
#else
__device__ __forceinline__ void copy_b128(const bf16* g, bf16* l) {
    *(bf16x8*)l = *(const bf16x8*)g;     // sync fallback: global load + ds_store
}
#  define WAIT_ASYNC(n) ((void)0)
#endif

// ---------------- WMMA fragment helpers (bf16 16x16x32) ----------------
// A (16x32, MxK): lanes 0-15 hold row M=lane, K = {0..7,16..23};
//                 lanes 16-31 hold row M=lane-16, K = {8..15,24..31}.
// B (32x16, KxN): lanes 0-15 hold col N=lane, K = {0..7,16..23};
//                 lanes 16-31 hold col N=lane-16, K = {8..15,24..31}.
// C/D (16x16 f32): VGPR r, lane l -> M = r + 8*(l/16), N = l%16.

__device__ __forceinline__ v16bf frag_cat(bf16x8 lo, bf16x8 hi) {
    return __builtin_shufflevector(lo, hi, 0,1,2,3,4,5,6,7,8,9,10,11,12,13,14,15);
}
__device__ __forceinline__ v16bf load_a_frag(const bf16* base, int row, int stride, int lane) {
    const bf16* p = base + row * stride + ((lane < 16) ? 0 : 8);
    return frag_cat(*(const bf16x8*)p, *(const bf16x8*)(p + 16));
}
__device__ __forceinline__ v16bf load_b_frag(const bf16* base, int col, int stride, int lane) {
    const bf16* p = base + col * stride + ((lane < 16) ? 0 : 8);
    return frag_cat(*(const bf16x8*)p, *(const bf16x8*)(p + 16));
}
__device__ __forceinline__ v8f wmma_bf16(v16bf a, v16bf b, v8f c) {
    return __builtin_amdgcn_wmma_f32_16x16x32_bf16(false, a, false, b, (short)0, c, false, false);
}

// ---------------- Generic GEMM: out = X[M,K] @ W[N,K]^T + bias ----------------
// mode 0/1: Q/K -> RoPE, store [B,H,S,D];  mode 2: V -> [B,H,D,S];  mode 3: plain [M,N]
#define BM 128
#define BN 128
#define BK 32
#define LDT 48          // padded LDS row stride (elements); 96B, 16B-aligned
#define ATILE (BM*LDT)  // per-buffer tile size in elements
#define GEMM_NIT (HIDDEN / BK)

__global__ __launch_bounds__(256)
void qkvo_gemm(const bf16* __restrict__ X, const bf16* __restrict__ W,
               const bf16* __restrict__ bias, bf16* __restrict__ out, int mode)
{
    __shared__ __attribute__((aligned(16))) bf16 As[2 * ATILE];   // double buffered
    __shared__ __attribute__((aligned(16))) bf16 Bs[2 * ATILE];

    const int tid  = threadIdx.x;
    const int lane = tid & 31;
    const int wv   = tid >> 5;        // 8 waves
    const int wr   = wv >> 1;         // 4 wave-rows (32 M each)
    const int wc   = wv & 1;          // 2 wave-cols (64 N each)
    const int rowBase = blockIdx.y * BM;
    const int colBase = blockIdx.x * BN;

    v8f acc[2][4] = {};

    // cooperative copy assignment: each thread moves 32B of A and 32B of B per tile
    const int cr   = tid >> 1;
    const int coff = (tid & 1) * 16;
    const bf16* xsrc = X + (size_t)(rowBase + cr) * HIDDEN + coff;
    const bf16* wsrc = W + (size_t)(colBase + cr) * HIDDEN + coff;
    bf16* adst = As + cr * LDT + coff;
    bf16* bdst = Bs + cr * LDT + coff;

    auto issue = [&](int it) {
        const int buf = (it & 1) * ATILE;
        const int k0  = it * BK;
        copy_b128(xsrc + k0,     adst + buf);
        copy_b128(xsrc + k0 + 8, adst + buf + 8);
        copy_b128(wsrc + k0,     bdst + buf);
        copy_b128(wsrc + k0 + 8, bdst + buf + 8);
    };

    issue(0);
    for (int it = 0; it < GEMM_NIT; ++it) {
        __syncthreads();                      // everyone done reading the buffer we refill
        if (it + 1 < GEMM_NIT) { issue(it + 1); WAIT_ASYNC(4); }  // 4 new ops stay in flight
        else                   { WAIT_ASYNC(0); }
        __syncthreads();                      // tile `it` visible to all waves

        const bf16* Ab = As + (it & 1) * ATILE;
        const bf16* Bb = Bs + (it & 1) * ATILE;
        v16bf a[2], b[4];
#pragma unroll
        for (int mi = 0; mi < 2; ++mi)
            a[mi] = load_a_frag(Ab, wr * 32 + mi * 16 + (lane & 15), LDT, lane);
#pragma unroll
        for (int nj = 0; nj < 4; ++nj)
            b[nj] = load_b_frag(Bb, wc * 64 + nj * 16 + (lane & 15), LDT, lane);
#pragma unroll
        for (int mi = 0; mi < 2; ++mi)
#pragma unroll
            for (int nj = 0; nj < 4; ++nj)
                acc[mi][nj] = wmma_bf16(a[mi], b[nj], acc[mi][nj]);
    }

    // ---------------- epilogue ----------------
    const int mhalf = (lane >> 4) * 8;
    const int ncol  = lane & 15;

#pragma unroll
    for (int mi = 0; mi < 2; ++mi) {
        const int rbase = rowBase + wr * 32 + mi * 16 + mhalf;
        if (mode == 3) {
#pragma unroll
            for (int nj = 0; nj < 4; ++nj) {
                const int col = colBase + wc * 64 + nj * 16 + ncol;
                const float bv = (float)bias[col];
#pragma unroll
                for (int r = 0; r < 8; ++r)
                    out[(size_t)(rbase + r) * HIDDEN + col] = (bf16)(acc[mi][nj][r] + bv);
            }
        } else if (mode == 2) {
            // V transposed [B,H,D,S]: lane's 8 rows = 8 consecutive s -> one b128 store
            const int bb = rbase >> 12, s = rbase & (SEQ - 1);
#pragma unroll
            for (int nj = 0; nj < 4; ++nj) {
                const int col = colBase + wc * 64 + nj * 16 + ncol; // h*64 + d
                const int h = col >> 6, d = col & 63;
                const float bv = (float)bias[col];
                bf16x8 pack;
#pragma unroll
                for (int r = 0; r < 8; ++r) pack[r] = (bf16)(acc[mi][nj][r] + bv);
                *(bf16x8*)(out + ((size_t)(bb * NUM_HEADS + h) * HEAD_SIZE + d) * SEQ + s) = pack;
            }
        } else {
            // Q/K with fused RoPE; wave N-span (64) is head-aligned: nj in {0,1} is d<32,
            // partner d+32 is fragment nj+2, same lane.
            const int bb = rbase >> 12;
#pragma unroll
            for (int nj = 0; nj < 2; ++nj) {
                const int col = colBase + wc * 64 + nj * 16 + ncol;
                const int h = col >> 6, d = col & 63;           // d in [0,32)
                const float b1 = (float)bias[col];
                const float b2 = (float)bias[col + 32];
                const float freq = __powf(10000.0f, -(float)d * (1.0f / 32.0f));
                const size_t obase = ((size_t)(bb * NUM_HEADS + h) * SEQ) * HEAD_SIZE;
#pragma unroll
                for (int r = 0; r < 8; ++r) {
                    const int s = (rbase + r) & (SEQ - 1);
                    float sn, cs;
                    __sincosf((float)s * freq, &sn, &cs);
                    const float x1 = acc[mi][nj][r]     + b1;   // q[d]
                    const float x2 = acc[mi][nj + 2][r] + b2;   // q[d+32]
                    bf16* op = out + obase + (size_t)s * HEAD_SIZE;
                    op[d]      = (bf16)(x1 * cs - x2 * sn);
                    op[d + 32] = (bf16)(x2 * cs + x1 * sn);
                }
            }
        }
    }
}

// ---- Flash attention: Q[BH,S,D] x K[BH,S,D] x Vt[BH,D,S] -> O[B,S,H*D] ----
#define KT 32          // key tile
#define LDK 72         // Ks row stride (elems), 144B
#define LDV 40         // Vs row stride (elems), 80B
#define KTILE (KT*LDK)
#define VTILE (HEAD_SIZE*LDV)
#define ATT_NIT (SEQ / KT)

__global__ __launch_bounds__(256)
void flash_attn(const bf16* __restrict__ Q, const bf16* __restrict__ K,
                const bf16* __restrict__ Vt, bf16* __restrict__ O)
{
    __shared__ __attribute__((aligned(16))) bf16 Ks[2 * KTILE];   // [key][d], dbl-buffered
    __shared__ __attribute__((aligned(16))) bf16 Vs[2 * VTILE];   // [d][key], dbl-buffered
    __shared__ __attribute__((aligned(16))) bf16 Ps[8 * 16 * 32]; // per-wave P tile

    const int tid  = threadIdx.x;
    const int lane = tid & 31;
    const int wv   = tid >> 5;
    const int blk  = blockIdx.x;
    const int qt   = blk & 31;           // 32 q-tiles of 128 per (b,h)
    const int bh   = blk >> 5;
    const int bb   = bh >> 4, h = bh & 15;

    const size_t qkb = (size_t)bh * SEQ * HEAD_SIZE;
    const size_t vtb = (size_t)bh * HEAD_SIZE * SEQ;
    const int qrow0  = qt * 128 + wv * 16;

    // cooperative copy assignment: 16B of K and 16B of Vt per thread per tile
    const bf16* ksrc = K  + qkb + (size_t)(tid >> 3) * HEAD_SIZE + (tid & 7) * 8;
    bf16*       kdst = Ks + (tid >> 3) * LDK + (tid & 7) * 8;
    const bf16* vsrc = Vt + vtb + (size_t)(tid >> 2) * SEQ + (tid & 3) * 8;
    bf16*       vdst = Vs + (tid >> 2) * LDV + (tid & 3) * 8;

    auto issue = [&](int it) {
        const int kt0 = it * KT;
        copy_b128(ksrc + (size_t)kt0 * HEAD_SIZE, kdst + (it & 1) * KTILE);
        copy_b128(vsrc + kt0,                     vdst + (it & 1) * VTILE);
    };

    // Q fragments (A layout), K-dim blocks d=[0,32) and [32,64), loaded once
    v16bf aQ[2];
    {
        const bf16* qp = Q + qkb + (size_t)(qrow0 + (lane & 15)) * HEAD_SIZE;
        const int off = (lane < 16) ? 0 : 8;
#pragma unroll
        for (int kb = 0; kb < 2; ++kb)
            aQ[kb] = frag_cat(*(const bf16x8*)(qp + kb * 32 + off),
                              *(const bf16x8*)(qp + kb * 32 + off + 16));
    }

    v8f o[4] = {};
    float m[8], l[8];
#pragma unroll
    for (int r = 0; r < 8; ++r) { m[r] = -1e30f; l[r] = 0.0f; }
    const float scale = 0.125f;   // 1/sqrt(64)

    issue(0);
    for (int it = 0; it < ATT_NIT; ++it) {
        __syncthreads();
        if (it + 1 < ATT_NIT) { issue(it + 1); WAIT_ASYNC(2); }   // 2 new ops in flight
        else                  { WAIT_ASYNC(0); }
        __syncthreads();

        const bf16* Kb = Ks + (it & 1) * KTILE;
        const bf16* Vb = Vs + (it & 1) * VTILE;

        // scores: 16 q-rows x 32 keys, K-dim = head 64 (2 wmma steps)
        v8f s0 = {}, s1 = {};
#pragma unroll
        for (int ks = 0; ks < 2; ++ks) {
            s0 = wmma_bf16(aQ[ks], load_b_frag(Kb + ks * 32,      (lane & 15), LDK, lane), s0);
            s1 = wmma_bf16(aQ[ks], load_b_frag(Kb + ks * 32, 16 + (lane & 15), LDK, lane), s1);
        }

        // online softmax (rows m<8 in lanes 0-15, m>=8 in 16-31; xor 1/2/4/8 stays in group)
        float corr[8];
#pragma unroll
        for (int r = 0; r < 8; ++r) {
            float a0 = s0[r] * scale, a1 = s1[r] * scale;
            float mx = fmaxf(a0, a1);
            mx = fmaxf(mx, __shfl_xor(mx, 1));
            mx = fmaxf(mx, __shfl_xor(mx, 2));
            mx = fmaxf(mx, __shfl_xor(mx, 4));
            mx = fmaxf(mx, __shfl_xor(mx, 8));
            const float nm = fmaxf(m[r], mx);
            corr[r] = __expf(m[r] - nm);
            const float p0 = __expf(a0 - nm);
            const float p1 = __expf(a1 - nm);
            s0[r] = p0; s1[r] = p1;
            float ps = p0 + p1;
            ps += __shfl_xor(ps, 1);
            ps += __shfl_xor(ps, 2);
            ps += __shfl_xor(ps, 4);
            ps += __shfl_xor(ps, 8);
            l[r] = l[r] * corr[r] + ps;
            m[r] = nm;
        }
#pragma unroll
        for (int nj = 0; nj < 4; ++nj)
#pragma unroll
            for (int r = 0; r < 8; ++r) o[nj][r] *= corr[r];

        // P: C-layout -> A-layout via wave-private LDS round trip
        bf16* pw = Ps + wv * (16 * 32);
        const int mloc = (lane >> 4) * 8, nloc = lane & 15;
#pragma unroll
        for (int r = 0; r < 8; ++r) {
            pw[(mloc + r) * 32 + nloc]      = (bf16)s0[r];
            pw[(mloc + r) * 32 + nloc + 16] = (bf16)s1[r];
        }
        v16bf aP;
        {
            const bf16* p = pw + (lane & 15) * 32 + ((lane < 16) ? 0 : 8);
            aP = frag_cat(*(const bf16x8*)p, *(const bf16x8*)(p + 16));
        }
        // O += P (16x32) @ V (32x64)
#pragma unroll
        for (int nj = 0; nj < 4; ++nj)
            o[nj] = wmma_bf16(aP, load_b_frag(Vb, nj * 16 + (lane & 15), LDV, lane), o[nj]);
    }

    // normalize + store O[B,S,H*D]
    const int mloc = (lane >> 4) * 8, nloc = lane & 15;
#pragma unroll
    for (int nj = 0; nj < 4; ++nj)
#pragma unroll
        for (int r = 0; r < 8; ++r) {
            const int s = qrow0 + mloc + r;
            O[((size_t)bb * SEQ + s) * HIDDEN + h * HEAD_SIZE + nj * 16 + nloc] =
                (bf16)(o[nj][r] / l[r]);
        }
}

// ---------------- host ----------------
extern "C" void kernel_launch(void* const* d_in, const int* in_sizes, int n_in,
                              void* d_out, int out_size, void* d_ws, size_t ws_size,
                              hipStream_t stream) {
    const bf16* x  = (const bf16*)d_in[0];
    const bf16* Wq = (const bf16*)d_in[1];
    const bf16* bq = (const bf16*)d_in[2];
    const bf16* Wk = (const bf16*)d_in[3];
    const bf16* bk = (const bf16*)d_in[4];
    const bf16* Wv = (const bf16*)d_in[5];
    const bf16* bv = (const bf16*)d_in[6];
    const bf16* Wo = (const bf16*)d_in[7];
    const bf16* bo = (const bf16*)d_in[8];
    bf16* out = (bf16*)d_out;

    const size_t elems = (size_t)BATCH * NUM_HEADS * SEQ * HEAD_SIZE; // 8M elems, 16MB
    bf16* Qb = (bf16*)d_ws;
    bf16* Kb = Qb + elems;
    bf16* Vb = Kb + elems;
    bf16* Ob = Vb + elems;

    dim3 blockDims(256);
    dim3 gemmGrid(HIDDEN / BN, MTOT / BM);                 // (8, 64)
    qkvo_gemm<<<gemmGrid, blockDims, 0, stream>>>(x,  Wq, bq, Qb, 0);
    qkvo_gemm<<<gemmGrid, blockDims, 0, stream>>>(x,  Wk, bk, Kb, 1);
    qkvo_gemm<<<gemmGrid, blockDims, 0, stream>>>(x,  Wv, bv, Vb, 2);
    flash_attn<<<dim3(BATCH * NUM_HEADS * (SEQ / 128)), blockDims, 0, stream>>>(Qb, Kb, Vb, Ob);
    qkvo_gemm<<<gemmGrid, blockDims, 0, stream>>>(Ob, Wo, bo, out, 3);

    (void)in_sizes; (void)n_in; (void)out_size; (void)ws_size;
}
